// SE3StructEncoder_35235911696634
// MI455X (gfx1250) — compile-verified, matrix operand order checked by reading, not run on hardware
//
#include <hip/hip_runtime.h>

#define H 128
#define L 512
#define B 2
#define NLAY 3
#define NSPLIT 2

typedef __attribute__((ext_vector_type(16))) __bf16 v16bf;
typedef __attribute__((ext_vector_type(8)))  float  v8f;

// Fast silu: v_exp + v_rcp + v_mul instead of the IEEE div chain.
__device__ __forceinline__ float silu_f(float x) {
  return x * __builtin_amdgcn_rcpf(1.0f + __expf(-x));
}

__device__ __forceinline__ unsigned ballot32(bool p) {
#if __has_builtin(__builtin_amdgcn_ballot_w32)
  return __builtin_amdgcn_ballot_w32(p);
#else
  return (unsigned)__ballot(p);
#endif
}

// ---------------------------------------------------------------------------
// Pack e_w2 (NLAY, H, H) f32 -> bf16 B-fragments in per-lane WMMA order.
// Layout: [lay][n_tile(8)][k_step(4)][lane(32)][elem(16)]
// elem e = 2v+p -> VGPR v, k = 32*ks + 8*half + 2*(v&3) + 16*(v>>2) + p
// n = 16*t + (lane&15)
// ---------------------------------------------------------------------------
__global__ void k_pack(const float* __restrict__ e_w2, __bf16* __restrict__ out) {
  int idx = blockIdx.x * blockDim.x + threadIdx.x;
  if (idx >= NLAY * 8 * 4 * 32 * 16) return;
  int e    = idx & 15;
  int rest = idx >> 4;
  int lane = rest & 31; rest >>= 5;
  int ks   = rest & 3;  rest >>= 2;
  int t    = rest & 7;  rest >>= 3;
  int lay  = rest;
  int half = lane >> 4;
  int v = e >> 1, p = e & 1;
  int n = t * 16 + (lane & 15);
  int k = ks * 32 + 8 * half + 2 * (v & 3) + ((v >> 2) << 4) + p;
  out[idx] = (__bf16)e_w2[(size_t)lay * H * H + k * H + n];
}

// ---------------------------------------------------------------------------
// Per-(b,i) neighbor compaction: idx[bi][0..cnt) = { j : contacts[bi][j] > 0 }.
// One wave per row; ballot + popcount prefix keeps order deterministic.
// ---------------------------------------------------------------------------
__global__ __launch_bounds__(32) void k_gather(const float* __restrict__ contacts,
                                               unsigned short* __restrict__ idx,
                                               int* __restrict__ cnt) {
  const int bi = blockIdx.x;
  const int lane = threadIdx.x;
  int base = 0;
  for (int c0 = 0; c0 < L; c0 += 32) {
    const int j = c0 + lane;
    const bool p = contacts[(size_t)bi * L + j] > 0.0f;
    const unsigned m = ballot32(p);
    if (p) {
      const int off = base + __popc(m & ((1u << lane) - 1u));
      idx[(size_t)bi * L + off] = (unsigned short)j;
    }
    base += __popc(m);
  }
  if (lane == 0) cnt[bi] = base;
}

// ---------------------------------------------------------------------------
// h = node_feats @ in_w + in_b    (B*L rows, IN_DIM=32 -> H)
// ---------------------------------------------------------------------------
__global__ void k_inproj(const float* __restrict__ nf, const float* __restrict__ w,
                         const float* __restrict__ b, float* __restrict__ h) {
  int row = blockIdx.x, c = threadIdx.x;
  const float* x = nf + row * 32;
  float acc = b[c];
#pragma unroll
  for (int k = 0; k < 32; ++k) acc += x[k] * w[k * H + c];
  h[row * H + c] = acc;
}

// ---------------------------------------------------------------------------
// hi = h @ w1[:H], hj = h @ w1[H:2H]
// ---------------------------------------------------------------------------
__global__ void k_proj2(const float* __restrict__ h, const float* __restrict__ w1,
                        float* __restrict__ hi, float* __restrict__ hj) {
  int row = blockIdx.x, c = threadIdx.x;
  __shared__ float xs[H];
  xs[c] = h[row * H + c];
  __syncthreads();
  float a = 0.f, bb = 0.f;
  for (int k = 0; k < H; ++k) {
    float xv = xs[k];
    a  += xv * w1[k * H + c];
    bb += xv * w1[(H + k) * H + c];
  }
  hi[row * H + c] = a;
  hj[row * H + c] = bb;
}

// ---------------------------------------------------------------------------
// Sparse edge MLP + aggregation via bf16 WMMA over gathered neighbor tiles.
// Contacted edges have contact == 1.0, so w_c folds into the bias vector.
// ---------------------------------------------------------------------------
__global__ __launch_bounds__(32) void edge_wmma_kernel(
    const float* __restrict__ hi_part, const float* __restrict__ hj_part,
    const float* __restrict__ coords,
    const unsigned short* __restrict__ nbr_idx, const int* __restrict__ nbr_cnt,
    const float* __restrict__ w1,      const float* __restrict__ b1,
    const v16bf* __restrict__ wpack,   const float* __restrict__ b2,
    float* __restrict__ m_out) {
  const int bi   = blockIdx.x;       // b*L + i
  const int b    = bi >> 9;
  const int lane = threadIdx.x;
  const int half = lane >> 4;
  const int r    = lane & 15;

  const int cnt_i = nbr_cnt[bi];
  if (cnt_i == 0) return;
  const int ntiles = (cnt_i + 15) >> 4;

  __shared__ float s_hic[H], s_wd[H];
  {
    const float* hirow = hi_part + (size_t)bi * H;
    const float* wd = w1 + 2 * H * H;        // row 2H   (dist weight)
    const float* wc = w1 + (2 * H + 1) * H;  // row 2H+1 (contact weight, ct==1)
    for (int k = lane; k < H; k += 32) {
      s_hic[k] = hirow[k] + b1[k] + wc[k];
      s_wd[k]  = wd[k];
    }
  }
  __syncthreads();

  const float cx = coords[bi * 3 + 0];
  const float cy = coords[bi * 3 + 1];
  const float cz = coords[bi * 3 + 2];

  // loop-invariant epilogue bias for this lane's column in each n-tile
  float b2v[8];
#pragma unroll
  for (int t = 0; t < 8; ++t) b2v[t] = b2[t * 16 + r];

  float m_acc[8];
#pragma unroll
  for (int t = 0; t < 8; ++t) m_acc[t] = 0.0f;

  const unsigned short* idxrow = nbr_idx + (size_t)bi * L;

#pragma clang loop unroll(disable)
  for (int tile = blockIdx.y; tile < ntiles; tile += NSPLIT) {
    const int slot  = tile * 16 + r;
    const bool valid = slot < cnt_i;
    const int jj = idxrow[valid ? slot : (cnt_i - 1)];
    const unsigned rowmask = ballot32((lane < 16) && valid);

    const float dx = cx - coords[(b * L + jj) * 3 + 0];
    const float dy = cy - coords[(b * L + jj) * 3 + 1];
    const float dz = cz - coords[(b * L + jj) * 3 + 2];
    const float sq = dx * dx + dy * dy + dz * dz;
    const float dist = sq > 0.0f ? sqrtf(sq) : 0.0f;

    const float* hjrow = hj_part + (size_t)(b * L + jj) * H;

    // prefetch next tile's gathered hj row (global_prefetch_b8)
    {
      const int pslot = slot + 16 * NSPLIT;
      const int pj = idxrow[pslot < cnt_i ? pslot : (cnt_i - 1)];
      __builtin_prefetch(hj_part + (size_t)(b * L + pj) * H, 0, 1);
    }

    // Build 4 A fragments: silu(pre) in 16-bit A layout (16x32 per step)
    v16bf afrag[4];
#pragma unroll
    for (int ks = 0; ks < 4; ++ks) {
      const int kb = ks * 32 + 8 * half;
#pragma unroll
      for (int v = 0; v < 8; ++v) {
        const int k = kb + 2 * (v & 3) + ((v >> 2) << 4);
        const float2 hj2 = *(const float2*)(hjrow + k);
        const float p0 = s_hic[k]     + hj2.x + dist * s_wd[k];
        const float p1 = s_hic[k + 1] + hj2.y + dist * s_wd[k + 1];
        afrag[ks][2 * v]     = (__bf16)silu_f(p0);
        afrag[ks][2 * v + 1] = (__bf16)silu_f(p1);
      }
    }

#pragma unroll
    for (int t = 0; t < 8; ++t) {
      v8f acc = {0.f, 0.f, 0.f, 0.f, 0.f, 0.f, 0.f, 0.f};
#pragma unroll
      for (int ks = 0; ks < 4; ++ks) {
        const v16bf bf = wpack[(t * 4 + ks) * 32 + lane];
        acc = __builtin_amdgcn_wmma_f32_16x16x32_bf16(
            false, afrag[ks], false, bf, (short)0, acc, false, false);
      }
      // epilogue: +b2, silu, mask tail rows, column reduce
      float colsum = 0.0f;
#pragma unroll
      for (int g = 0; g < 8; ++g) {
        const int row = g + 8 * half;     // D layout: VGPR g -> M=g (+8 upper half)
        const float x = acc[g] + b2v[t];
        const float s = silu_f(x);
        if ((rowmask >> row) & 1u) colsum += s;
      }
      colsum += __shfl_xor(colsum, 16, 32);  // combine M=0..7 with M=8..15
      m_acc[t] += colsum;
    }
  }

  if (lane < 16) {
#pragma unroll
    for (int t = 0; t < 8; ++t)
      atomicAdd(m_out + (size_t)bi * H + t * 16 + lane, m_acc[t]);
  }
}

// ---------------------------------------------------------------------------
// tmp = silu(h @ n_w1[:H] + m @ n_w1[H:] + n_b1)
// ---------------------------------------------------------------------------
__global__ void k_node1(const float* __restrict__ h, const float* __restrict__ m,
                        const float* __restrict__ w1, const float* __restrict__ b1,
                        float* __restrict__ tmp) {
  int row = blockIdx.x, c = threadIdx.x;
  __shared__ float hs[H], ms[H];
  hs[c] = h[row * H + c];
  ms[c] = m[row * H + c];
  __syncthreads();
  float acc = b1[c];
  for (int k = 0; k < H; ++k)
    acc += hs[k] * w1[k * H + c] + ms[k] * w1[(H + k) * H + c];
  tmp[row * H + c] = silu_f(acc);
}

// ---------------------------------------------------------------------------
// out = h + tmp @ n_w2 + n_b2
// ---------------------------------------------------------------------------
__global__ void k_node2(const float* __restrict__ h, const float* __restrict__ tmp,
                        const float* __restrict__ w2, const float* __restrict__ b2,
                        float* __restrict__ out) {
  int row = blockIdx.x, c = threadIdx.x;
  __shared__ float ts[H];
  ts[c] = tmp[row * H + c];
  __syncthreads();
  float acc = b2[c];
  for (int k = 0; k < H; ++k) acc += ts[k] * w2[k * H + c];
  out[row * H + c] = h[row * H + c] + acc;
}

extern "C" void kernel_launch(void* const* d_in, const int* in_sizes, int n_in,
                              void* d_out, int out_size, void* d_ws, size_t ws_size,
                              hipStream_t stream) {
  (void)in_sizes; (void)n_in; (void)out_size; (void)ws_size;
  const float* coords   = (const float*)d_in[0];
  const float* contacts = (const float*)d_in[1];
  const float* node_fe  = (const float*)d_in[2];
  const float* in_w     = (const float*)d_in[3];
  const float* in_b     = (const float*)d_in[4];
  const float* e_w1     = (const float*)d_in[5];
  const float* e_b1     = (const float*)d_in[6];
  const float* e_w2     = (const float*)d_in[7];
  const float* e_b2     = (const float*)d_in[8];
  const float* n_w1     = (const float*)d_in[9];
  const float* n_b1     = (const float*)d_in[10];
  const float* n_w2     = (const float*)d_in[11];
  const float* n_b2     = (const float*)d_in[12];
  float* out = (float*)d_out;

  char* ws = (char*)d_ws;
  const size_t NBL = (size_t)B * L * H * sizeof(float);  // 512 KB
  float* h    = (float*)(ws + 0 * NBL);
  float* hi   = (float*)(ws + 1 * NBL);
  float* hj   = (float*)(ws + 2 * NBL);
  float* mi   = (float*)(ws + 3 * NBL);
  float* tmp  = (float*)(ws + 4 * NBL);
  v16bf* pack = (v16bf*)(ws + 5 * NBL);                     // 96 KB bf16 B frags
  unsigned short* nidx = (unsigned short*)(ws + 5 * NBL + 102400);  // 1 MB
  int* ncnt = (int*)(ws + 5 * NBL + 102400 + (size_t)B * L * L * 2); // 4 KB

  k_pack<<<192, 256, 0, stream>>>(e_w2, (__bf16*)pack);
  k_gather<<<B * L, 32, 0, stream>>>(contacts, nidx, ncnt);
  k_inproj<<<B * L, H, 0, stream>>>(node_fe, in_w, in_b, h);

  for (int lay = 0; lay < NLAY; ++lay) {
    const float* w1 = e_w1 + (size_t)lay * (2 * H + 2) * H;
    k_proj2<<<B * L, H, 0, stream>>>(h, w1, hi, hj);
    hipMemsetAsync(mi, 0, NBL, stream);
    edge_wmma_kernel<<<dim3(B * L, NSPLIT), 32, 0, stream>>>(
        hi, hj, coords, nidx, ncnt, w1, e_b1 + lay * H,
        pack + (size_t)lay * 8 * 4 * 32, e_b2 + lay * H, mi);
    k_node1<<<B * L, H, 0, stream>>>(h, mi, n_w1 + (size_t)lay * 2 * H * H,
                                     n_b1 + lay * H, tmp);
    k_node2<<<B * L, H, 0, stream>>>(h, tmp, n_w2 + (size_t)lay * H * H,
                                     n_b2 + lay * H, (lay == NLAY - 1) ? out : h);
  }
}
